// FlashAttentionFusedAttention_4587025072520
// MI455X (gfx1250) — matrix operand order repaired and checked by reading, not compiled, not run on hardware
//
#include <hip/hip_runtime.h>
#include <hip/hip_bf16.h>

// ---------------------------------------------------------------------------
// Fused GQA attention block for MI455X (gfx1250), wave32 + WMMA bf16.
//   B=2, S=2048, HID=2048, H=16, HKV=8, D=128, NREP=2
// Pipeline: gemm<0> (x@wq + RoPE -> Q bf16), gemm<1> (x@wkv + RoPE(K) -> K,V),
//           flash attention (WMMA QK^T and P@V, online softmax),
//           gemm<2> (att @ wo -> f32 d_out).
// GEMM: 128x256 block tile, 64x64 wave tile -> 16 WMMAs per wave per K-step.
// Pure-copy LDS stagings use GLOBAL_LOAD_ASYNC_TO_LDS_B128 (ASYNCcnt-tracked).
// Workspace layout (48 MB): Q 16MB | K 8MB | V 8MB | attout 16MB.
// ---------------------------------------------------------------------------

typedef __attribute__((ext_vector_type(16))) __bf16 v16bf;
typedef __attribute__((ext_vector_type(8)))  __bf16 v8bf;
typedef __attribute__((ext_vector_type(4)))  __bf16 v4bf;
typedef __attribute__((ext_vector_type(2)))  __bf16 v2bf;
typedef __attribute__((ext_vector_type(8)))  float  v8f;
typedef __attribute__((ext_vector_type(4)))  int    v4i;

#define WMMA_BF16(a, b, c) \
  __builtin_amdgcn_wmma_f32_16x16x32_bf16(false, (a), false, (b), (short)0, (c), false, false)

#if __has_builtin(__builtin_amdgcn_global_load_async_to_lds_b128)
#define HAVE_ASYNC_LDS 1
#endif

#define GAS __attribute__((address_space(1)))
#define LAS __attribute__((address_space(3)))

static constexpr int Bc = 2, Sc = 2048, HIDc = 2048, Hc = 16, HKVc = 8, Dc = 128;

// 16-byte global->LDS copy: async (ASYNCcnt-tracked, no VGPR round trip) when
// the toolchain has the gfx1250 builtin, else a plain b128 copy.
__device__ inline void async_cp16(__bf16* l, const __bf16* g) {
#ifdef HAVE_ASYNC_LDS
  __builtin_amdgcn_global_load_async_to_lds_b128(
      (GAS v4i*)(void*)g, (LAS v4i*)(void*)l, 0, 0);
#else
  *(uint4*)l = *(const uint4*)g;
#endif
}

__device__ inline void wait_async_zero() {
#ifdef HAVE_ASYNC_LDS
#if __has_builtin(__builtin_amdgcn_s_wait_asynccnt)
  __builtin_amdgcn_s_wait_asynccnt(0);
#else
  asm volatile("s_wait_asynccnt 0x0" ::: "memory");
#endif
#endif
}

__device__ inline v8f v8f_zero() {
  v8f z;
#pragma unroll
  for (int i = 0; i < 8; ++i) z[i] = 0.0f;
  return z;
}

struct frag_bits { uint4 lo, hi; };

// Build a v16bf fragment from two 16-byte chunks (LDS or global).
__device__ inline v16bf frag_ld(const __bf16* a, const __bf16* b) {
  frag_bits p;
  p.lo = *(const uint4*)a;
  p.hi = *(const uint4*)b;
  return __builtin_bit_cast(v16bf, p);
}

// ---------------------------------------------------------------------------
// GEMM, block tile 128(M) x 256(N), K-step 32, 8 waves (2x4), wave tile 64x64.
// MODE 0: A=f32 x, W=wq, epilogue RoPE all cols -> Q bf16 (B,H,S,D)
// MODE 1: A=f32 x, W=wkv, epilogue RoPE cols<1024 -> K bf16, cols>=1024 -> V
// MODE 2: A=bf16 attout, W=wo, epilogue plain f32 store to d_out
// LDS: As row-major (M x K); Bt transposed (N x K) so B-frags are contiguous.
// ---------------------------------------------------------------------------
template <int MODE>
__global__ __launch_bounds__(256) void gemm_kernel(
    const void* __restrict__ Aptr, const float* __restrict__ W,
    const float* __restrict__ fcos, const float* __restrict__ fsin,
    __bf16* __restrict__ qout, __bf16* __restrict__ kout,
    __bf16* __restrict__ vout, float* __restrict__ fout) {
  const int tid = threadIdx.x;
  const int wave = tid >> 5;
  const int lane = tid & 31;
  const int l15 = lane & 15;
  const int hi = lane >> 4;          // 0 = lanes 0-15, 1 = lanes 16-31
  const int m0 = blockIdx.y * 128;
  const int n0 = blockIdx.x * 256;
  const int wm = wave >> 2;          // 0..1 : 64-row slice
  const int wn = wave & 3;           // 0..3 : 64-col slice

  __shared__ __align__(16) __bf16 As[128 * 32];   // (m, k) row-major, 8 KB
  __shared__ __align__(16) __bf16 Bt[256 * 32];   // (n, k) transposed, 16 KB

  v8f acc[4][4];
#pragma unroll
  for (int mt = 0; mt < 4; ++mt)
#pragma unroll
    for (int nt = 0; nt < 4; ++nt) acc[mt][nt] = v8f_zero();

  for (int k0 = 0; k0 < HIDc; k0 += 32) {
    __syncthreads();
    // ---- stage A tile (128x32) into LDS as bf16 ----
    if (MODE == 2) {
      const __bf16* A = (const __bf16*)Aptr;
      const int r = tid >> 1;               // 0..127
      const int c = (tid & 1) * 16;
      const __bf16* src = A + (size_t)(m0 + r) * HIDc + k0 + c;
      async_cp16(&As[r * 32 + c], src);          // global_load_async_to_lds
      async_cp16(&As[r * 32 + c + 8], src + 8);
    } else {
      const float* A = (const float*)Aptr;
#pragma unroll
      for (int p = 0; p < 4; ++p) {
        const int r = (tid >> 3) + p * 32;
        const int c = (tid & 7) * 4;
        const float4 v = *(const float4*)(A + (size_t)(m0 + r) * HIDc + k0 + c);
        v4bf t;
        t[0] = (__bf16)v.x; t[1] = (__bf16)v.y;
        t[2] = (__bf16)v.z; t[3] = (__bf16)v.w;
        *(v4bf*)&As[r * 32 + c] = t;        // ds_store_b64
      }
    }
    // ---- stage W tile (32x256) transposed into Bt (coalesced row reads) ----
    {
      const float* wcol = W + (size_t)k0 * HIDc + n0 + tid;   // one column/thread
#pragma unroll
      for (int g = 0; g < 4; ++g) {
        v8bf t;
#pragma unroll
        for (int kk = 0; kk < 8; ++kk)
          t[kk] = (__bf16)wcol[(size_t)(g * 8 + kk) * HIDc];
        *(v8bf*)&Bt[tid * 32 + g * 8] = t;  // ds_store_b128
      }
    }
    if (k0 + 32 < HIDc)  // hint next weight tile toward the caches
      __builtin_prefetch(W + (size_t)(k0 + 32) * HIDc + n0 + tid, 0, 1);
    if (MODE == 2) wait_async_zero();
    __syncthreads();

    // ---- fragments: everything is two 16B vector loads ----
    const int kbA = hi * 8;    // A-frag K base per lane group
    const int kbB = hi * 8;    // B-frag K base per lane group
    v16bf bfrag[4];
#pragma unroll
    for (int nt = 0; nt < 4; ++nt) {
      const int n = wn * 64 + nt * 16 + l15;
      bfrag[nt] = frag_ld(&Bt[n * 32 + kbB], &Bt[n * 32 + kbB + 16]);
    }
#pragma unroll
    for (int mt = 0; mt < 4; ++mt) {
      const int row = wm * 64 + mt * 16 + l15;
      const v16bf afrag = frag_ld(&As[row * 32 + kbA], &As[row * 32 + kbA + 16]);
#pragma unroll
      for (int nt = 0; nt < 4; ++nt)
        acc[mt][nt] = WMMA_BF16(afrag, bfrag[nt], acc[mt][nt]);
    }
  }

  // ---- epilogue: C layout vgpr=row(+8 for hi lanes), lane=col ----
#pragma unroll
  for (int mt = 0; mt < 4; ++mt) {
#pragma unroll
    for (int nt = 0; nt < 4; ++nt) {
      const int ncol = n0 + wn * 64 + nt * 16 + l15;
#pragma unroll
      for (int r = 0; r < 8; ++r) {
        const int m = m0 + wm * 64 + mt * 16 + r + hi * 8;
        float v = acc[mt][nt][r];
        if (MODE == 2) {
          fout[(size_t)m * HIDc + ncol] = v;
        } else {
          const int b = m >> 11;      // m / S
          const int s = m & 2047;     // m % S
          const float partner = __shfl_xor(v, 1);  // RoPE pair (2i, 2i+1)
          const bool doRope = (MODE == 0) || (ncol < HKVc * Dc);
          if (doRope) {
            const int fi = (ncol & (Dc - 1)) >> 1;
            const float cs = fcos[s * (Dc / 2) + fi];
            const float sn = fsin[s * (Dc / 2) + fi];
            v = ((ncol & 1) == 0) ? (v * cs - partner * sn)
                                  : (v * cs + partner * sn);
          }
          const __bf16 o = (__bf16)v;
          if (MODE == 0) {
            const int h = ncol >> 7, d = ncol & (Dc - 1);
            qout[(((size_t)b * Hc + h) * Sc + s) * Dc + d] = o;
          } else {
            if (ncol < HKVc * Dc) {
              const int h = ncol >> 7, d = ncol & (Dc - 1);
              kout[(((size_t)b * HKVc + h) * Sc + s) * Dc + d] = o;
            } else {
              const int c2 = ncol - HKVc * Dc;
              const int h = c2 >> 7, d = c2 & (Dc - 1);
              vout[(((size_t)b * HKVc + h) * Sc + s) * Dc + d] = o;
            }
          }
        }
      }
    }
  }
}

// ---------------------------------------------------------------------------
// Flash attention: 8 waves x 16 query rows = 128-row Q tile per block,
// 32-key tiles staged in LDS, online softmax, WMMA for QK^T and P@V.
// Ks kept (key, d) row-major (score B-frags read along d = contiguous);
// Vs stored transposed (d, key) so P@V B-frags read along key = contiguous.
// K staging uses async global->LDS; V staging transposes through VGPRs.
// grid = (S/128, B*H)
// ---------------------------------------------------------------------------
__global__ __launch_bounds__(256) void flash_kernel(
    const __bf16* __restrict__ Q, const __bf16* __restrict__ K,
    const __bf16* __restrict__ V, const int* __restrict__ seqlens,
    __bf16* __restrict__ attout) {
  const int tid = threadIdx.x;
  const int wave = tid >> 5;
  const int lane = tid & 31;
  const int l15 = lane & 15;
  const int hi = lane >> 4;
  const int bh = blockIdx.y;
  const int b = bh >> 4;         // / H
  const int h = bh & (Hc - 1);   // % H
  const int hk = h >> 1;         // NREP = 2
  const int q0 = blockIdx.x * 128;
  const int qr = q0 + wave * 16;
  const int slen = seqlens[b];

  __shared__ __align__(16) __bf16 Ks[32 * 128];      // (key, d)
  __shared__ __align__(16) __bf16 Vt[128 * 32];      // (d, key) transposed
  __shared__ __align__(16) __bf16 Ps[8 * 16 * 32];   // per-wave 16x32 P

  const __bf16* Qp = Q + ((size_t)b * Hc + h) * Sc * Dc;
  const __bf16* Kp = K + ((size_t)b * HKVc + hk) * Sc * Dc;
  const __bf16* Vp = V + ((size_t)b * HKVc + hk) * Sc * Dc;

  const int kbA = hi * 8;
  const int kbB = hi * 8;

  // Q held in registers: 4 A-fragments covering D=128, vector global loads
  v16bf aq[4];
  {
    const __bf16* qrow = Qp + (size_t)(qr + l15) * Dc;
#pragma unroll
    for (int c = 0; c < 4; ++c)
      aq[c] = frag_ld(qrow + c * 32 + kbA, qrow + c * 32 + kbA + 16);
  }

  v8f accO[8];
#pragma unroll
  for (int i = 0; i < 8; ++i) accO[i] = v8f_zero();
  float mrow[8], lrow[8];
#pragma unroll
  for (int r = 0; r < 8; ++r) { mrow[r] = -1e30f; lrow[r] = 0.0f; }

  const float scale = 0.08838834764831845f;  // 1/sqrt(128)
  const int nkt = (q0 + 128) >> 5;           // causal: keys <= q0+127

  // staging indices (hoisted)
  const int kr = tid >> 3;            // K tile row 0..31
  const int kc = (tid & 7) * 16;      // K tile d chunk
  const int vr2 = tid >> 4;           // V: key pair index 0..15
  const int vd8 = (tid & 15) * 8;     // V: d chunk

  for (int kt = 0; kt < nkt; ++kt) {
    const int k0 = kt * 32;
    __syncthreads();
    // ---- stage K via async global->LDS, V transposed via VGPRs ----
    {
      const __bf16* ks = Kp + (size_t)(k0 + kr) * Dc + kc;
      async_cp16(&Ks[kr * 128 + kc], ks);
      async_cp16(&Ks[kr * 128 + kc + 8], ks + 8);
      // V: read two key rows, write (d, key) pairs as 4-byte stores
      const v8bf va = *(const v8bf*)(Vp + (size_t)(k0 + 2 * vr2) * Dc + vd8);
      const v8bf vb = *(const v8bf*)(Vp + (size_t)(k0 + 2 * vr2 + 1) * Dc + vd8);
#pragma unroll
      for (int j = 0; j < 8; ++j) {
        v2bf pr;
        pr[0] = va[j];
        pr[1] = vb[j];
        *(v2bf*)&Vt[(vd8 + j) * 32 + 2 * vr2] = pr;   // ds_store_b32
      }
    }
    wait_async_zero();
    __syncthreads();

    // ---- S = Q K^T : two 16-key column tiles, 4 WMMAs each over D ----
    v8f sf[2];
    sf[0] = v8f_zero();
    sf[1] = v8f_zero();
#pragma unroll
    for (int ct = 0; ct < 2; ++ct) {
      const int n = ct * 16 + l15;   // local key (B-fragment column)
#pragma unroll
      for (int c = 0; c < 4; ++c) {
        const v16bf bk = frag_ld(&Ks[n * 128 + c * 32 + kbB],
                                 &Ks[n * 128 + c * 32 + kbB + 16]);
        sf[ct] = WMMA_BF16(aq[c], bk, sf[ct]);
      }
    }

    // ---- scale + causal/pad mask ----
    float pm[2][8];
#pragma unroll
    for (int ct = 0; ct < 2; ++ct) {
      const int key = k0 + ct * 16 + l15;
#pragma unroll
      for (int r = 0; r < 8; ++r) {
        const int qrow = qr + r + hi * 8;
        float v = sf[ct][r] * scale;
        if (key > qrow || key >= slen) v = -1e30f;
        pm[ct][r] = v;
      }
    }

    // ---- online softmax per row (16-lane shuffle reductions) ----
#pragma unroll
    for (int r = 0; r < 8; ++r) {
      float mx = fmaxf(pm[0][r], pm[1][r]);
#pragma unroll
      for (int off = 8; off >= 1; off >>= 1) mx = fmaxf(mx, __shfl_xor(mx, off));
      const float mnew = fmaxf(mrow[r], mx);
      const float p0 = __expf(pm[0][r] - mnew);
      const float p1 = __expf(pm[1][r] - mnew);
      float rs = p0 + p1;
#pragma unroll
      for (int off = 8; off >= 1; off >>= 1) rs += __shfl_xor(rs, off);
      const float corr = __expf(mrow[r] - mnew);
      lrow[r] = lrow[r] * corr + rs;
      mrow[r] = mnew;
#pragma unroll
      for (int dt = 0; dt < 8; ++dt) accO[dt][r] *= corr;
      const int rl = r + hi * 8;  // C-layout row -> P row
      Ps[wave * 512 + rl * 32 + l15]      = (__bf16)p0;
      Ps[wave * 512 + rl * 32 + 16 + l15] = (__bf16)p1;
    }

    // ---- O += P V ----
    const v16bf ap = frag_ld(&Ps[wave * 512 + l15 * 32 + kbA],
                             &Ps[wave * 512 + l15 * 32 + kbA + 16]);
#pragma unroll
    for (int dt = 0; dt < 8; ++dt) {
      const int n = dt * 16 + l15;
      const v16bf bv = frag_ld(&Vt[n * 32 + kbB], &Vt[n * 32 + kbB + 16]);
      accO[dt] = WMMA_BF16(ap, bv, accO[dt]);
    }
  }

  // ---- finalize: divide by l, store bf16 att-out (B, S, H*D) ----
  float invl[8];
#pragma unroll
  for (int r = 0; r < 8; ++r) invl[r] = 1.0f / lrow[r];
#pragma unroll
  for (int dt = 0; dt < 8; ++dt) {
    const int dcol = dt * 16 + l15;
#pragma unroll
    for (int r = 0; r < 8; ++r) {
      const int sq = qr + r + hi * 8;
      attout[((size_t)b * Sc + sq) * (Hc * Dc) + h * Dc + dcol] =
          (__bf16)(accO[dt][r] * invl[r]);
    }
  }
}

// ---------------------------------------------------------------------------
extern "C" void kernel_launch(void* const* d_in, const int* in_sizes, int n_in,
                              void* d_out, int out_size, void* d_ws, size_t ws_size,
                              hipStream_t stream) {
  const float* x    = (const float*)d_in[0];
  const float* wq   = (const float*)d_in[1];
  const float* wkv  = (const float*)d_in[2];
  const float* wo   = (const float*)d_in[3];
  const float* fcos = (const float*)d_in[4];
  const float* fsin = (const float*)d_in[5];
  const int*   seql = (const int*)d_in[6];

  char* ws = (char*)d_ws;
  __bf16* Qbf = (__bf16*)(ws);                          // B*H*S*D   bf16 = 16 MB
  __bf16* Kbf = (__bf16*)(ws + (16u << 20));            // B*HKV*S*D bf16 =  8 MB
  __bf16* Vbf = (__bf16*)(ws + (24u << 20));            // B*HKV*S*D bf16 =  8 MB
  __bf16* Obf = (__bf16*)(ws + (32u << 20));            // B*S*H*D   bf16 = 16 MB

  const dim3 blk(256);
  const dim3 gemm_grid(HIDc / 256, (Bc * Sc) / 128);    // (8, 32)
  const dim3 fa_grid(Sc / 128, Bc * Hc);                // (16, 32)

  gemm_kernel<0><<<gemm_grid, blk, 0, stream>>>(x, wq, fcos, fsin,
                                                Qbf, nullptr, nullptr, nullptr);
  gemm_kernel<1><<<gemm_grid, blk, 0, stream>>>(x, wkv, fcos, fsin,
                                                nullptr, Kbf, Vbf, nullptr);
  flash_kernel<<<fa_grid, blk, 0, stream>>>(Qbf, Kbf, Vbf, seql, Obf);
  gemm_kernel<2><<<gemm_grid, blk, 0, stream>>>(Obf, wo, nullptr, nullptr,
                                                nullptr, nullptr, nullptr,
                                                (float*)d_out);
}
